// Block_7258494730369
// MI455X (gfx1250) — matrix operand order repaired and checked by reading, not compiled, not run on hardware
//
#include <hip/hip_runtime.h>
#include <cstdint>
#include <cstddef>

typedef unsigned short u16;
typedef __attribute__((ext_vector_type(16))) __bf16 v16bf;
typedef __attribute__((ext_vector_type(8)))  __bf16 v8bf;
typedef __attribute__((ext_vector_type(8)))  float  v8f;

// ---------------- constants for this problem ----------------
#define BB  4
#define TT  1024
#define CC  1024
#define HH  16
#define HD  64
#define FF  4096
#define BT  (BB*TT)   // 4096

// Try the CDNA5 async global->LDS DMA path (CDNA5 ISA).
// Flip to 0 if the assembler rejects the mnemonic.
#define USE_ASYNC_LDS 1

// ---------------- helpers ----------------
__device__ __forceinline__ u16 f2bf(float f) {
  union { float f; uint32_t u; } c; c.f = f;
  return (u16)((c.u + 0x7FFFu + ((c.u >> 16) & 1u)) >> 16);
}

__device__ __forceinline__ v16bf cat8(v8bf lo, v8bf hi) {
  v16bf r;
#pragma unroll
  for (int i = 0; i < 8; ++i) { r[i] = lo[i]; r[i + 8] = hi[i]; }
  return r;
}

// A-fragment (16x32 bf16, MxK) from row-major memory. Lane l holds row m=l&15.
// Contiguous layout: elems e -> K = 16h + e, h = l>>4.
__device__ __forceinline__ v16bf load_a_frag(const u16* base, int lda) {
  int l = threadIdx.x & 31;
  const u16* p = base + (size_t)(l & 15) * lda + ((l >> 4) << 4);
  v8bf lo = *(const v8bf*)p;
  v8bf hi = *(const v8bf*)(p + 8);
  return cat8(lo, hi);
}

// 16 contiguous bf16 per lane. B-fragment when memory holds B^T rows per lane:
// lane n = l&15, elems e -> K = 16*(l>>4) + e  (caller computes the lane pointer).
__device__ __forceinline__ v16bf load16(const u16* p) {
  v8bf lo = *(const v8bf*)p;
  v8bf hi = *(const v8bf*)(p + 8);
  return cat8(lo, hi);
}

__device__ __forceinline__ v8f wmma_bf16(v16bf a, v16bf b, v8f c) {
  return __builtin_amdgcn_wmma_f32_16x16x32_bf16(false, a, false, b, (short)0, c,
                                                 false, false);
}

__device__ __forceinline__ v8f zero8() {
  return (v8f){0.f,0.f,0.f,0.f,0.f,0.f,0.f,0.f};
}

// 16B global -> LDS stage. Async DMA path avoids the VGPR bounce.
__device__ __forceinline__ void stage_b128(const u16* g, u16* lds) {
#if USE_ASYNC_LDS
  asm volatile("global_load_async_to_lds_b128 %0, %1, off"
               :: "v"((uint32_t)(uintptr_t)lds), "v"(g)
               : "memory");
#else
  *(uint4*)lds = *(const uint4*)g;
#endif
}
__device__ __forceinline__ void stage_wait() {
#if USE_ASYNC_LDS
  asm volatile("s_wait_asynccnt 0x0" ::: "memory");
#endif
}

// ---------------- LayerNorm: fp32 row -> bf16 row ----------------
__global__ void ln_kernel(const float* __restrict__ X, const float* __restrict__ g,
                          const float* __restrict__ bta, u16* __restrict__ Out) {
  int row = blockIdx.x; int tid = threadIdx.x;
  const float* xr = X + (size_t)row * CC;
  float v0 = xr[tid], v1 = xr[tid + 256], v2 = xr[tid + 512], v3 = xr[tid + 768];
  __shared__ float red[256];
  red[tid] = v0 + v1 + v2 + v3;
  __syncthreads();
  for (int o = 128; o > 0; o >>= 1) { if (tid < o) red[tid] += red[tid + o]; __syncthreads(); }
  float mu = red[0] * (1.0f / CC);
  __syncthreads();
  float d0 = v0 - mu, d1 = v1 - mu, d2 = v2 - mu, d3 = v3 - mu;
  red[tid] = d0 * d0 + d1 * d1 + d2 * d2 + d3 * d3;
  __syncthreads();
  for (int o = 128; o > 0; o >>= 1) { if (tid < o) red[tid] += red[tid + o]; __syncthreads(); }
  float rstd = rsqrtf(red[0] * (1.0f / CC) + 1e-5f);
  u16* orow = Out + (size_t)row * CC;
  orow[tid]       = f2bf(d0 * rstd * g[tid]       + bta[tid]);
  orow[tid + 256] = f2bf(d1 * rstd * g[tid + 256] + bta[tid + 256]);
  orow[tid + 512] = f2bf(d2 * rstd * g[tid + 512] + bta[tid + 512]);
  orow[tid + 768] = f2bf(d3 * rstd * g[tid + 768] + bta[tid + 768]);
}

// ---------------- Weight packing: fp32 [K,N] (head-strided) -> bf16 B-fragment tiles
// out[tile*512 + l*16 + e] = W[k = kt*32 + 16*(l>>4) + e][n = nt*16 + (l&15)]
__global__ void pack_b_kernel(const float* __restrict__ W, u16* __restrict__ Out,
                              int K, int N, int strideK, int headN, long strideHead) {
  long idx = (long)blockIdx.x * 256 + threadIdx.x;
  long tile = idx >> 9; int within = (int)(idx & 511);
  int l = within >> 4, e = within & 15;
  int ntilesN = N >> 4;
  long kt = tile / ntilesN; int nt = (int)(tile % ntilesN);
  long k = kt * 32 + ((l >> 4) << 4) + e;
  int  n = nt * 16 + (l & 15);
  long src = (long)(n / headN) * strideHead + k * (long)strideK + (n % headN);
  Out[idx] = f2bf(W[src]);
}

// ---------------- GEMM: bf16 A [M,K] x packed-bf16 B [K,N] (+bias, epilogue) -------
// MODE 0: out bf16 = acc + bias ; MODE 1: out bf16 = gelu(acc + bias)
// MODE 2: out fp32 = acc + bias + resid
// 256 threads (8 waves), tile 64x128; wave (mt=w&3, nq=w>>2): 16x64 strip,
// A-frag reused 4x per K step -> 4 WMMA / 160B loaded per wave-step.
template <int MODE>
__global__ void gemm_kernel(const u16* __restrict__ A, const u16* __restrict__ Bp,
                            const float* __restrict__ bias, const float* __restrict__ resid,
                            void* __restrict__ Out, int M, int N, int K) {
  int tid = threadIdx.x; int w = tid >> 5; int l = tid & 31;
  int ln = l & 15, lh = l >> 4;
  int mt = w & 3, nq = w >> 2;
  int gm0 = blockIdx.x * 64 + mt * 16;
  int n0  = blockIdx.y * 128 + nq * 64;
  int ntilesN = N >> 4;
  int nt0 = n0 >> 4;

  v8f acc[4];
#pragma unroll
  for (int j = 0; j < 4; ++j) acc[j] = zero8();

  const u16* Abase = A + (size_t)gm0 * K;
  for (int kt = 0; kt < (K >> 5); ++kt) {
    v16bf a = load_a_frag(Abase + kt * 32, K);
#pragma unroll
    for (int j = 0; j < 4; ++j) {
      const u16* bp = Bp + (((size_t)kt * ntilesN + nt0 + j) << 9) + (size_t)l * 16;
      v16bf b = load16(bp);
      acc[j] = wmma_bf16(a, b, acc[j]);
    }
  }

#pragma unroll
  for (int j = 0; j < 4; ++j) {
    int gn = n0 + j * 16 + ln;
    float bv = bias ? bias[gn] : 0.0f;
#pragma unroll
    for (int r = 0; r < 8; ++r) {
      int gm = gm0 + r + 8 * lh;
      float v = acc[j][r] + bv;
      size_t idx = (size_t)gm * N + gn;
      if (MODE == 0) {
        ((u16*)Out)[idx] = f2bf(v);
      } else if (MODE == 1) {
        ((u16*)Out)[idx] = f2bf(0.5f * v * (1.0f + erff(v * 0.70710678118f)));
      } else {
        ((float*)Out)[idx] = v + resid[idx];
      }
    }
  }
}

// ---------------- Flash-style causal attention (S^T formulation) ----------------
// Q,K,V,O: bf16 [B*T, H*HD] row-major (col = head*64 + d).
// Grid: (T/64, B*H), block 128 (4 waves); wave w owns query rows qbase..qbase+15.
// We compute S^T = K_tile(16x64) x Q^T(64x16): lane = query column, so softmax
// stats are register-resident (+one shfl_xor(16)), and exp(S) packs DIRECTLY
// into the PV A-fragment (K-map e<8 -> kh0 r=e, e>=8 -> kh1 r=e-8). No P
// transpose through LDS. K tiles stream via async global->LDS; V is staged
// transposed so PV B-frags are contiguous ds_load_b128s.
__global__ void attn_kernel(const u16* __restrict__ Q, const u16* __restrict__ Km,
                            const u16* __restrict__ Vm, u16* __restrict__ O) {
  int bh = blockIdx.y; int b = bh >> 4; int hd = bh & 15;
  int tid = threadIdx.x; int w = tid >> 5; int l = tid & 31;
  int ln = l & 15, lh = l >> 4;
  int qbase = blockIdx.x * 64 + w * 16;

  __shared__ u16 sK[32 * 64];      // [key][d] row-major
  __shared__ u16 sVt[64 * 32];     // [d][s]  (transposed V tile)
  __shared__ float sStat[4][16];   // per-wave per-query rescale / rowsum

  // Q^T B-fragments (lane ln = query, elems -> d = dt*32 + 16*lh + e), from global once
  const u16* qp = Q + (size_t)(b * TT + qbase + ln) * CC + hd * 64 + lh * 16;
  v16bf bq0 = load16(qp);        // d = 0..31
  v16bf bq1 = load16(qp + 32);   // d = 32..63

  float rmax = -1e30f, rsum = 0.0f;
  v8f oa[4];
#pragma unroll
  for (int j = 0; j < 4; ++j) oa[j] = zero8();
  int q = qbase + ln;

  int nst = (blockIdx.x + 1) * 2;  // uniform across block (causal bound)
  for (int st = 0; st < nst; ++st) {
    int s0 = st * 32;
    { // cooperative staging: 32B of K (async DMA) + 32B of V (transposed) per thread
      int r = tid >> 2, cs = (tid & 3) << 4;
      const u16* gk = Km + (size_t)(b * TT + s0 + r) * CC + hd * 64 + cs;
      stage_b128(gk,     sK + r * 64 + cs);
      stage_b128(gk + 8, sK + r * 64 + cs + 8);
      const uint4* gv = (const uint4*)(Vm + (size_t)(b * TT + s0 + r) * CC + hd * 64 + cs);
      uint4 t0 = gv[0], t1 = gv[1];
      const u16* e0 = (const u16*)&t0;
      const u16* e1 = (const u16*)&t1;
#pragma unroll
      for (int i = 0; i < 8; ++i) sVt[(cs + i) * 32 + r]     = e0[i];
#pragma unroll
      for (int i = 0; i < 8; ++i) sVt[(cs + 8 + i) * 32 + r] = e1[i];
    }
    stage_wait();
    __syncthreads();

    // S^T: two 16-key subtiles; lane holds query q, regs r -> key s0+kh*16+r+8*lh
    float p[2][8];
    float mnew = rmax;
#pragma unroll
    for (int kh = 0; kh < 2; ++kh) {
      v16bf ak0 = load_a_frag(sK + kh * 16 * 64, 64);       // d 0..31
      v16bf ak1 = load_a_frag(sK + kh * 16 * 64 + 32, 64);  // d 32..63
      v8f s = zero8();
      s = wmma_bf16(ak0, bq0, s);
      s = wmma_bf16(ak1, bq1, s);
#pragma unroll
      for (int r = 0; r < 8; ++r) {
        int sg = s0 + kh * 16 + r + 8 * lh;
        float v = s[r] * 0.125f;
        p[kh][r] = (sg > q) ? -1e30f : v;
        mnew = fmaxf(mnew, p[kh][r]);
      }
    }
    mnew = fmaxf(mnew, __shfl_xor(mnew, 16, 32));

    // exp + pack straight into the PV A-fragment
    float ps = 0.0f;
    v16bf pa;
#pragma unroll
    for (int kh = 0; kh < 2; ++kh) {
#pragma unroll
      for (int r = 0; r < 8; ++r) {
        float e = __expf(p[kh][r] - mnew);
        ps += e;
        pa[kh * 8 + r] = (__bf16)e;
      }
    }
    ps += __shfl_xor(ps, 16, 32);
    float co = __expf(rmax - mnew);
    rsum = rsum * co + ps;
    rmax = mnew;
    sStat[w][ln] = co;            // both halves write identical value
    __syncthreads();

    // rescale O rows (query = r + 8*lh in O layout) and accumulate P*V
    float cr[8];
#pragma unroll
    for (int r = 0; r < 8; ++r) cr[r] = sStat[w][r + 8 * lh];
#pragma unroll
    for (int j = 0; j < 4; ++j) {
      v16bf bv = cat8(*(const v8bf*)(sVt + (size_t)(j * 16 + ln) * 32 + lh * 8), *(const v8bf*)(sVt + (size_t)(j * 16 + ln) * 32 + 16 + lh * 8));
#pragma unroll
      for (int r = 0; r < 8; ++r) oa[j][r] *= cr[r];
      oa[j] = wmma_bf16(pa, bv, oa[j]);
    }
    __syncthreads();
  }

  // normalize + write out (concat-head layout)
  sStat[w][ln] = rsum;
  __syncthreads();
  float inv[8];
#pragma unroll
  for (int r = 0; r < 8; ++r) inv[r] = 1.0f / sStat[w][r + 8 * lh];
#pragma unroll
  for (int j = 0; j < 4; ++j) {
#pragma unroll
    for (int r = 0; r < 8; ++r) {
      float val = oa[j][r] * inv[r];
      O[(size_t)(b * TT + qbase + r + 8 * lh) * CC + hd * 64 + j * 16 + ln] = f2bf(val);
    }
  }
}

// ---------------- host orchestration ----------------
extern "C" void kernel_launch(void* const* d_in, const int* in_sizes, int n_in,
                              void* d_out, int out_size, void* d_ws, size_t ws_size,
                              hipStream_t stream) {
  (void)in_sizes; (void)n_in; (void)out_size; (void)ws_size;
  const float* x     = (const float*)d_in[0];
  const float* ln1_g = (const float*)d_in[1];
  const float* ln1_b = (const float*)d_in[2];
  const float* ln2_g = (const float*)d_in[3];
  const float* ln2_b = (const float*)d_in[4];
  const float* Wq    = (const float*)d_in[5];
  const float* Wk    = (const float*)d_in[6];
  const float* Wv    = (const float*)d_in[7];
  const float* Wproj = (const float*)d_in[8];
  const float* bproj = (const float*)d_in[9];
  const float* W1    = (const float*)d_in[10];
  const float* b1    = (const float*)d_in[11];
  const float* W2    = (const float*)d_in[12];
  const float* b2    = (const float*)d_in[13];
  float* out = (float*)d_out;

  char* ws = (char*)d_ws;
  size_t off = 0;
  auto alloc = [&](size_t bytes) -> char* {
    char* p = ws + off;
    off = (off + bytes + 255) & ~(size_t)255;
    return p;
  };
  u16*   xn1  = (u16*)  alloc((size_t)BT * CC * 2);
  u16*   qb   = (u16*)  alloc((size_t)BT * CC * 2);
  u16*   kb   = (u16*)  alloc((size_t)BT * CC * 2);
  u16*   vb   = (u16*)  alloc((size_t)BT * CC * 2);
  u16*   attb = (u16*)  alloc((size_t)BT * CC * 2);
  float* x1   = (float*)alloc((size_t)BT * CC * 4);
  u16*   xn2  = (u16*)  alloc((size_t)BT * CC * 2);
  u16*   hbuf = (u16*)  alloc((size_t)BT * FF * 2);
  u16*   Wq_p = (u16*)  alloc((size_t)CC * CC * 2);
  u16*   Wk_p = (u16*)  alloc((size_t)CC * CC * 2);
  u16*   Wv_p = (u16*)  alloc((size_t)CC * CC * 2);
  u16*   Wp_p = (u16*)  alloc((size_t)CC * CC * 2);
  u16*   W1_p = (u16*)  alloc((size_t)CC * FF * 2);
  u16*   W2_p = (u16*)  alloc((size_t)FF * CC * 2);

  // LN1
  ln_kernel<<<BT, 256, 0, stream>>>(x, ln1_g, ln1_b, xn1);

  // pack weights into B-fragment layout
  int pkCC = (CC * CC) / 256, pkCF = (CC * FF) / 256;
  pack_b_kernel<<<pkCC, 256, 0, stream>>>(Wq, Wq_p, CC, CC, HD, HD, (long)CC * HD);
  pack_b_kernel<<<pkCC, 256, 0, stream>>>(Wk, Wk_p, CC, CC, HD, HD, (long)CC * HD);
  pack_b_kernel<<<pkCC, 256, 0, stream>>>(Wv, Wv_p, CC, CC, HD, HD, (long)CC * HD);
  pack_b_kernel<<<pkCC, 256, 0, stream>>>(Wproj, Wp_p, CC, CC, CC, CC, 0);
  pack_b_kernel<<<pkCF, 256, 0, stream>>>(W1, W1_p, CC, FF, FF, FF, 0);
  pack_b_kernel<<<pkCF, 256, 0, stream>>>(W2, W2_p, FF, CC, CC, CC, 0);

  // QKV projections (bf16 out, layout [b,t,h,d])
  dim3 gqkv(BT / 64, CC / 128);
  gemm_kernel<0><<<gqkv, 256, 0, stream>>>(xn1, Wq_p, nullptr, nullptr, qb, BT, CC, CC);
  gemm_kernel<0><<<gqkv, 256, 0, stream>>>(xn1, Wk_p, nullptr, nullptr, kb, BT, CC, CC);
  gemm_kernel<0><<<gqkv, 256, 0, stream>>>(xn1, Wv_p, nullptr, nullptr, vb, BT, CC, CC);

  // causal flash attention
  attn_kernel<<<dim3(TT / 64, BB * HH), 128, 0, stream>>>(qb, kb, vb, attb);

  // output projection + bias + residual -> x1 (fp32)
  gemm_kernel<2><<<dim3(BT / 64, CC / 128), 256, 0, stream>>>(attb, Wp_p, bproj, x, x1,
                                                              BT, CC, CC);
  // LN2
  ln_kernel<<<BT, 256, 0, stream>>>(x1, ln2_g, ln2_b, xn2);

  // FFN
  gemm_kernel<1><<<dim3(BT / 64, FF / 128), 256, 0, stream>>>(xn2, W1_p, b1, nullptr, hbuf,
                                                              BT, FF, CC);
  gemm_kernel<2><<<dim3(BT / 64, CC / 128), 256, 0, stream>>>(hbuf, W2_p, b2, x1, out,
                                                              BT, CC, FF);
}
